// CoarseMatcher_18674517803616
// MI455X (gfx1250) — matrix operand order repaired and checked by reading, not compiled, not run on hardware
//
#include <hip/hip_runtime.h>

// ---------------- static problem config (matches reference) ----------------
#define Bc   4
#define Lc   4800          // L == S == H0C*W0C
#define Cc   256
#define H0   60
#define W0   80
#define THRv 0.2f
#define SIMSCALE (1.0f/25.6f)   // 1/(C*TEMP) ; sim = dot/(C*TEMP)
#define SCALEc   8.0f

// ---------------- vector types for WMMA ----------------
typedef __attribute__((ext_vector_type(16))) __bf16 v16bf;
typedef __attribute__((ext_vector_type(8)))  __bf16 v8bf;
typedef __attribute__((ext_vector_type(4)))  __bf16 v4bf;
typedef __attribute__((ext_vector_type(8)))  float  v8f;

constexpr int BM = 128, BN = 128, BK = 32;
constexpr int NT = (Lc + BM - 1) / BM;     // 38 tiles per dim
constexpr int LDSS = 48;                   // bf16 elems per LDS row (32 + 16 pad, 96B: 16B aligned)

// Async global->LDS copy of 32 bytes per lane, no VGPR staging (ASYNCcnt).
// NOTE: IOFFSET is added to BOTH the global and LDS addresses (ISA §15.18.3),
// so one base pair + offset:16 covers the full 32B.
__device__ __forceinline__ void async_copy32(const void* g, void* lds) {
  unsigned       l  = (unsigned)(uintptr_t)lds;            // low 32b == LDS offset
  unsigned long long ga = (unsigned long long)(uintptr_t)g;
  asm volatile("global_load_async_to_lds_b128 %0, %1, off\n\t"
               "global_load_async_to_lds_b128 %0, %1, off offset:16"
               :: "v"(l), "v"(ga) : "memory");
}
__device__ __forceinline__ void wait_async0() {
  asm volatile("s_wait_asynccnt 0x0" ::: "memory");
}

// ---------------- K0: f32 -> bf16 convert ----------------
__global__ __launch_bounds__(256)
void cvt_kernel(const float* __restrict__ src, __bf16* __restrict__ dst, int n4) {
  int i = blockIdx.x * blockDim.x + threadIdx.x;
  if (i < n4) {
    float4 f = ((const float4*)src)[i];
    v4bf o = { (__bf16)f.x, (__bf16)f.y, (__bf16)f.z, (__bf16)f.w };
    ((v4bf*)dst)[i] = o;
  }
}

// ---------------- K1: bf16 WMMA GEMM -> E=exp(sim), fused row/col sums ------
__global__ __launch_bounds__(256)
void gemm_exp_kernel(const __bf16* __restrict__ f0, const __bf16* __restrict__ f1,
                     float* __restrict__ E,
                     float* __restrict__ rowSumG, float* __restrict__ colSumG) {
  __shared__ __bf16 shA[2][BM * LDSS];
  __shared__ __bf16 shB[2][BN * LDSS];
  __shared__ float  sRowSum[BM];
  __shared__ float  sColSum[BN];

  const int b   = blockIdx.z;
  const int bm  = blockIdx.y * BM;
  const int bn  = blockIdx.x * BN;
  const int tid = threadIdx.x;
  const int lane  = tid & 31;
  const int wave  = tid >> 5;     // 8 waves
  const int waveM = wave >> 1;    // 0..3 -> 32-row slab
  const int waveN = wave & 1;     // 0..1 -> 64-col slab
  const int lm = lane & 15;
  const int hf = lane >> 4;

  if (tid < BM) { sRowSum[tid] = 0.f; sColSum[tid] = 0.f; }

  // global->LDS async mapping: each thread moves 32B (16 bf16) per tile
  const int ldRow  = tid >> 1;
  const int ldHalf = tid & 1;
  const int gA = min(bm + ldRow, Lc - 1);
  const int gB = min(bn + ldRow, Lc - 1);
  const __bf16* srcA0 = f0 + ((size_t)(b * Lc + gA)) * Cc + ldHalf * 16;
  const __bf16* srcB0 = f1 + ((size_t)(b * Lc + gB)) * Cc + ldHalf * 16;
  __bf16* dstA0 = &shA[0][ldRow * LDSS + ldHalf * 16];
  __bf16* dstB0 = &shB[0][ldRow * LDSS + ldHalf * 16];
  __bf16* dstA1 = &shA[1][ldRow * LDSS + ldHalf * 16];
  __bf16* dstB1 = &shB[1][ldRow * LDSS + ldHalf * 16];

  // prologue: fill buffer 0 asynchronously (no VGPR staging)
  async_copy32(srcA0, dstA0);
  async_copy32(srcB0, dstB0);
  wait_async0();

  v8f acc[2][4] = {};

  __syncthreads();
  for (int kk = 0; kk < Cc / BK; ++kk) {
    const int cur = kk & 1;
    if (kk + 1 < Cc / BK) {       // async fill of the other buffer
      async_copy32(srcA0 + (kk + 1) * BK, cur ? dstA0 : dstA1);
      async_copy32(srcB0 + (kk + 1) * BK, cur ? dstB0 : dstB1);
    }
    if (kk + 2 < Cc / BK) {       // global_prefetch_b8 (L2 warm)
      __builtin_prefetch(srcA0 + (kk + 2) * BK, 0, 0);
      __builtin_prefetch(srcB0 + (kk + 2) * BK, 0, 0);
    }

    // A fragment: lane<16 holds K {0..7,16..23}, lane>=16 holds K {8..15,24..31}
    v16bf aF[2];
#pragma unroll
    for (int ti = 0; ti < 2; ++ti) {
      const __bf16* p = &shA[cur][(waveM * 32 + ti * 16 + lm) * LDSS + hf * 8];
      union { v16bf v; v8bf h[2]; } u;
      u.h[0] = *(const v8bf*)p;
      u.h[1] = *(const v8bf*)(p + 16);
      aF[ti] = u.v;
    }
    // B fragment: column n=lm, lane<16 holds K {0..7,16..23}, lane>=16 holds K {8..15,24..31}
    v16bf bF[4];
#pragma unroll
    for (int tj = 0; tj < 4; ++tj) {
      const __bf16* p = &shB[cur][(waveN * 64 + tj * 16 + lm) * LDSS + hf * 8];
      union { v16bf v; v8bf h[2]; } u;
      u.h[0] = *(const v8bf*)p;
      u.h[1] = *(const v8bf*)(p + 16);
      bF[tj] = u.v;
    }
#pragma unroll
    for (int ti = 0; ti < 2; ++ti)
#pragma unroll
      for (int tj = 0; tj < 4; ++tj)
        acc[ti][tj] = __builtin_amdgcn_wmma_f32_16x16x32_bf16(
            false, aF[ti], false, bF[tj], (short)0, acc[ti][tj], false, false);

    if (kk + 1 < Cc / BK) wait_async0();   // own ASYNCcnt drained ...
    __syncthreads();                        // ... barrier makes all fills visible
  }

  // Epilogue: E = exp(sim), fused row/col exp-sums (softmax denominators)
#pragma unroll
  for (int ti = 0; ti < 2; ++ti) {
#pragma unroll
    for (int tj = 0; tj < 4; ++tj) {
      const int colLoc = waveN * 64 + tj * 16 + lm;
      const int gcol   = bn + colLoc;
      float e[8];
      float csum = 0.f;
#pragma unroll
      for (int r = 0; r < 8; ++r) {
        const int rowLoc = waveM * 32 + ti * 16 + hf * 8 + r;
        const int grow   = bm + rowLoc;
        float ev = __expf(acc[ti][tj][r] * SIMSCALE);
        if (grow >= Lc || gcol >= Lc) ev = 0.f;
        else E[((size_t)(b * Lc + grow)) * Lc + gcol] = ev;
        e[r] = ev;
        csum += ev;
      }
      atomicAdd(&sColSum[colLoc], csum);          // ds_add_f32
#pragma unroll
      for (int r = 0; r < 8; ++r) {               // reduce over 16 lanes (one N-strip)
        float v = e[r];
        v += __shfl_xor(v, 1, 16);
        v += __shfl_xor(v, 2, 16);
        v += __shfl_xor(v, 4, 16);
        v += __shfl_xor(v, 8, 16);
        if (lm == 0) {
          const int rowLoc = waveM * 32 + ti * 16 + hf * 8 + r;
          atomicAdd(&sRowSum[rowLoc], v);
        }
      }
    }
  }
  __syncthreads();
  if (tid < BM) {
    const int gm = bm + tid;
    if (gm < Lc) atomicAdd(&rowSumG[b * Lc + gm], sRowSum[tid]);   // global_atomic_add_f32
    const int gn = bn + tid;
    if (gn < Lc) atomicAdd(&colSumG[b * Lc + gn], sColSum[tid]);
  }
}

// ---------------- K2: conf = E^2/(rowSum*colSum) in-place + col conf max ----
__global__ __launch_bounds__(256)
void conf_kernel(float* __restrict__ conf,
                 const float* __restrict__ rowSumG, const float* __restrict__ colSumG,
                 unsigned int* __restrict__ colCMaxG) {
  __shared__ float tile[128][132];     // 67.6 KB of the 320 KB WGP LDS
  __shared__ float sInvRow[128];
  __shared__ float sInvCol[128];
  __shared__ float part[256];

  const int b  = blockIdx.z;
  const int bm = blockIdx.y * 128;
  const int bn = blockIdx.x * 128;
  const int tid = threadIdx.x;

  if (tid < 128) {
    int gm = bm + tid; sInvRow[tid] = (gm < Lc) ? 1.f / rowSumG[b * Lc + gm] : 0.f;
    int gn = bn + tid; sInvCol[tid] = (gn < Lc) ? 1.f / colSumG[b * Lc + gn] : 0.f;
  }
  __syncthreads();

  const int row = tid >> 1;
  const int ch  = (tid & 1) * 64;
  const int gm  = bm + row;
  const float ir = sInvRow[row];

  if (bm + 128 <= Lc && bn + 128 <= Lc) {
    float4* ptr = (float4*)(conf + ((size_t)(b * Lc + gm)) * Lc + bn + ch);
#pragma unroll 4
    for (int j = 0; j < 16; ++j) {
      float4 e  = ptr[j];
      float4 ic = *(const float4*)&sInvCol[ch + 4 * j];
      float4 o;
      o.x = e.x * e.x * ir * ic.x;  o.y = e.y * e.y * ir * ic.y;
      o.z = e.z * e.z * ir * ic.z;  o.w = e.w * e.w * ir * ic.w;
      ptr[j] = o;
      *(float4*)&tile[row][ch + 4 * j] = o;
    }
  } else {
    for (int j = 0; j < 64; ++j) {
      const int gs = bn + ch + j;
      float o = 0.f;
      if (gm < Lc && gs < Lc) {
        const size_t idx = ((size_t)(b * Lc + gm)) * Lc + gs;
        const float e = conf[idx];
        o = e * e * ir * sInvCol[ch + j];
        conf[idx] = o;
      }
      tile[row][ch + j] = o;
    }
  }
  __syncthreads();

  // LDS transpose read: column maxima (conf >= 0 so int-bit max is order-safe)
  const int c2 = tid & 127;
  const int r0 = (tid >> 7) * 64;
  float cm = 0.f;
  for (int r = 0; r < 64; ++r) cm = fmaxf(cm, tile[r0 + r][c2]);
  part[tid] = cm;
  __syncthreads();
  if (tid < 128) {
    const float m = fmaxf(part[tid], part[tid + 128]);
    const int gn = bn + tid;
    if (gn < Lc) atomicMax(&colCMaxG[b * Lc + gn], __float_as_uint(m));
  }
}

// ---------------- K3: mask + mutual-NN argmax + outputs ----------------
__global__ __launch_bounds__(256)
void match_kernel(const float* __restrict__ conf, const unsigned int* __restrict__ colCMaxG,
                  float* __restrict__ oMatch, float* __restrict__ oSid,
                  float* __restrict__ oMconf, float* __restrict__ oK0,
                  float* __restrict__ oK1) {
  const int l = blockIdx.x, b = blockIdx.y, tid = threadIdx.x;
  const int i0 = l / W0, j0 = l - i0 * W0;
  const bool rowValid = (i0 >= 2 && i0 < H0 - 2 && j0 >= 2 && j0 < W0 - 2);

  __shared__ float bv[256];
  __shared__ int   bi[256];
  float best = -1.f; int bidx = 1 << 30;

  if (rowValid) {
    const float* rowp = conf + ((size_t)(b * Lc) + l) * Lc;
    for (int s = tid; s < Lc; s += 256) {
      const int i1 = s / W0, j1 = s - i1 * W0;
      if (i1 < 2 || i1 >= H0 - 2 || j1 < 2 || j1 >= W0 - 2) continue;
      const float c = rowp[s];
      if (c > THRv) {
        // score = mask * rowConfMax[l] * colConfMax[s]; rowConfMax is a
        // positive constant per row -> argmax over colConfMax only
        const float v = __uint_as_float(colCMaxG[b * Lc + s]);
        if (v > best) { best = v; bidx = s; }   // s increasing: first-max kept
      }
    }
  }
  bv[tid] = best; bi[tid] = bidx;
  __syncthreads();
  for (int st = 128; st > 0; st >>= 1) {
    if (tid < st) {
      if (bv[tid + st] > bv[tid] ||
          (bv[tid + st] == bv[tid] && bi[tid + st] < bi[tid])) {
        bv[tid] = bv[tid + st]; bi[tid] = bi[tid + st];
      }
    }
    __syncthreads();
  }
  if (tid == 0) {
    const bool found = bv[0] > 0.f;
    const int  sid   = found ? bi[0] : 0;
    const size_t o   = (size_t)b * Lc + l;
    oMatch[o] = found ? 1.f : 0.f;
    oSid[o]   = (float)sid;
    oMconf[o] = found ? conf[((size_t)(b * Lc) + l) * Lc + sid] : 0.f;
    const int si = sid / W0, sj = sid - si * W0;
    oK1[o * 2]     = sj * SCALEc;
    oK1[o * 2 + 1] = si * SCALEc;
    if (b == 0) { oK0[l * 2] = j0 * SCALEc; oK0[l * 2 + 1] = i0 * SCALEc; }
  }
}

// ---------------- host-side orchestration ----------------
extern "C" void kernel_launch(void* const* d_in, const int* in_sizes, int n_in,
                              void* d_out, int out_size, void* d_ws, size_t ws_size,
                              hipStream_t stream) {
  const float* feat0 = (const float*)d_in[0];
  const float* feat1 = (const float*)d_in[1];
  float* out = (float*)d_out;

  char* ws = (char*)d_ws;
  const size_t nFeat = (size_t)Bc * Lc * Cc;          // 4,915,200 elems
  __bf16* f0b = (__bf16*)ws;
  __bf16* f1b = (__bf16*)(ws + nFeat * 2);
  float*  rowSum = (float*)(ws + nFeat * 4);
  float*  colSum = rowSum + (size_t)Bc * Lc;
  unsigned int* colCMax = (unsigned int*)(colSum + (size_t)Bc * Lc);

  // zero the accumulators every call (graph-capturable memset node)
  hipMemsetAsync(rowSum, 0, 3 * (size_t)Bc * Lc * sizeof(float), stream);

  const int n4 = (int)(nFeat / 4);
  cvt_kernel<<<dim3((n4 + 255) / 256), 256, 0, stream>>>(feat0, f0b, n4);
  cvt_kernel<<<dim3((n4 + 255) / 256), 256, 0, stream>>>(feat1, f1b, n4);

  float* conf = out;                                   // E stored in-place, then conf
  dim3 g(NT, NT, Bc);
  gemm_exp_kernel<<<g, 256, 0, stream>>>(f0b, f1b, conf, rowSum, colSum);
  conf_kernel<<<g, 256, 0, stream>>>(conf, rowSum, colSum, colCMax);

  const size_t confN = (size_t)Bc * Lc * Lc;
  float* oMatch = out + confN;
  float* oSid   = oMatch + (size_t)Bc * Lc;
  float* oMconf = oSid   + (size_t)Bc * Lc;
  float* oK0    = oMconf + (size_t)Bc * Lc;
  float* oK1    = oK0    + (size_t)Lc * 2;
  match_kernel<<<dim3(Lc, Bc), 256, 0, stream>>>(conf, colCMax, oMatch, oSid,
                                                 oMconf, oK0, oK1);
}
